// DCNv2_137438953846
// MI455X (gfx1250) — matrix operand order repaired and checked by reading, not compiled, not run on hardware
//
#include <hip/hip_runtime.h>
#include <hip/hip_bf16.h>
#include <math.h>

// ---------------------------------------------------------------------------
// DCNv2 forward for MI455X (gfx1250, wave32, WMMA).
//   x:(4,256,64,64) f32, w_off:(27,256,3,3), b_off:(27), w_main:(256,256,3,3)
//   out:(4,256,64,64) f32
// Both convs are M=16384, K=2304 GEMMs on v_wmma_f32_16x16x32_f16.
// B matrices pre-swizzled to fragment order -> direct global v16h frag loads.
// A tiles built branchless in LDS (validity folded into bilinear weights),
// double-buffered, one barrier per K-chunk.
// ---------------------------------------------------------------------------

typedef __attribute__((ext_vector_type(16))) _Float16 v16h;
typedef __attribute__((ext_vector_type(8)))  _Float16 v8h;
typedef __attribute__((ext_vector_type(8)))  float    v8f;

#define NB    4
#define CIN   256
#define HW    64
#define OUTC  256
#define KK9   9
#define KTOT  (KK9 * CIN)          // 2304
#define NPIX  (NB * HW * HW)       // 16384
#define OCOFF 27

#define OM_FLOATS   ((size_t)NPIX * OCOFF)   // 442368
#define BMAIN_HALFS ((size_t)KTOT * OUTC)    // 589824 (72 kblk x 16 cgrp tiles)
#define BOFF_HALFS  ((size_t)KTOT * 32)      // 73728  (72 kblk x 2 cgrp tiles)
#define NCHUNK      72                       // 2304 / 32

// --------------------------------------------------------------------------
// Kernel 0: weights fp32 -> f16, swizzled to WMMA-B fragment order.
// Tile (kblk, cgrp) is 32 lanes x 16 contiguous halfs:
//   flat = ((kblk*NCG + cgrp)*32 + L)*16 + e
//   k = kblk*32 + (L>=16)*8 + (e&7) + (e>=8)*16 ; col = cgrp*16 + (L&15) ; k = tap*256 + c
// --------------------------------------------------------------------------
__global__ __launch_bounds__(256)
void prep_weights(const float* __restrict__ w_off,
                  const float* __restrict__ w_main,
                  _Float16* __restrict__ Bmain,
                  _Float16* __restrict__ Boff) {
  const int idx = blockIdx.x * 256 + threadIdx.x;
  if (idx < (int)BMAIN_HALFS) {
    const int e    = idx & 15;
    const int L    = (idx >> 4) & 31;
    const int cgrp = (idx >> 9) & 15;
    const int kblk = idx >> 13;
    const int k  = kblk * 32 + (L >> 4) * 8 + (e & 7) + (e >> 3) * 16;
    const int c  = k & 255, kk = k >> 8;
    const int o  = cgrp * 16 + (L & 15);
    Bmain[idx] = (_Float16)w_main[((size_t)o * CIN + c) * KK9 + kk];
  } else {
    const int j = idx - (int)BMAIN_HALFS;
    if (j < (int)BOFF_HALFS) {
      const int e    = j & 15;
      const int L    = (j >> 4) & 31;
      const int cgrp = (j >> 9) & 1;
      const int kblk = j >> 10;
      const int k  = kblk * 32 + (L >> 4) * 8 + (e & 7) + (e >> 3) * 16;
      const int c  = k & 255, kk = k >> 8;
      const int oc = cgrp * 16 + (L & 15);
      const float v = (oc < OCOFF)
                        ? w_off[((size_t)oc * CIN + c) * KK9 + kk] : 0.0f;
      Boff[j] = (_Float16)v;
    }
  }
}

// A fragment (16x32 f16): lane<16 holds K {0..7,16..23}, lane>=16 {8..15,24..31}
__device__ __forceinline__ v16h load_afrag(const _Float16* rowp, int akb) {
  const v8h a0 = *(const v8h*)(rowp + akb);        // ds_load_b128
  const v8h a1 = *(const v8h*)(rowp + akb + 16);   // ds_load_b128
  v16h af;
  #pragma unroll
  for (int e = 0; e < 8; ++e) { af[e] = a0[e]; af[8 + e] = a1[e]; }
  return af;
}

// --------------------------------------------------------------------------
// Kernel 1: 3x3 offset conv as implicit-GEMM WMMA.
// Block <-> (n, oh): 64 pixels (ow) x 32 outc. 8 waves = 4(M) x 2(N).
// --------------------------------------------------------------------------
__global__ __launch_bounds__(256)
void offset_conv(const float* __restrict__ x,
                 const _Float16* __restrict__ Boff,
                 const float* __restrict__ bias,
                 float* __restrict__ om) {
  __shared__ __align__(16) _Float16 Ald[2][64][40];  // double-buffered A tile

  const int tid  = threadIdx.x;
  const int lane = tid & 31, wv = tid >> 5;
  const int wm = wv >> 1, wn = wv & 1;
  const int blk = blockIdx.x;              // n*64 + oh
  const int n = blk >> 6, oh = blk & 63;
  const float* xn = x + (size_t)n * CIN * HW * HW;

  v8f acc = {};
  const int ai = tid >> 2;                 // pixel (== ow) 0..63
  const int cg = tid & 3;                  // channel group of 8
  const int ar  = wm * 16 + (lane & 15);
  const int akb = (lane >> 4) * 8;

  int chunk = 0;
  for (int ky = 0; ky < 3; ++ky) {
    const int iy  = oh + ky - 1;
    const int iyc = min(max(iy, 0), 63);
    for (int kx = 0; kx < 3; ++kx) {
      const int ix  = ai + kx - 1;
      const bool vok = ((unsigned)iy < 64u) && ((unsigned)ix < 64u);
      const float f  = vok ? 1.0f : 0.0f;
      const int base = iyc * HW + min(max(ix, 0), 63);
      for (int cc = 0; cc < 8; ++cc, ++chunk) {
        const int par = chunk & 1;
        // ---- build A tile (branchless, 1 b128 ds store/thread) ----
        const int cb = cc * 32 + cg * 8;
        union { _Float16 h[8]; uint4 u; } pk;
        #pragma unroll
        for (int j = 0; j < 8; ++j)
          pk.h[j] = (_Float16)(f * xn[(size_t)(cb + j) * (HW * HW) + base]);
        *(uint4*)&Ald[par][ai][cg * 8] = pk.u;
        __syncthreads();
        // ---- fragments + WMMA (B straight from global, fragment order) ----
        const v16h af = load_afrag(&Ald[par][ar][0], akb);
        const v16h bf = *(const v16h*)
            (Boff + ((size_t)(chunk * 2 + wn) * 32 + lane) * 16);
        acc = __builtin_amdgcn_wmma_f32_16x16x32_f16(false, af, false, bf,
                                                     (short)0, acc,
                                                     false, false);
      }
    }
  }
  // ---- epilogue: bias (+ sigmoid on mask channels 18..26) ----
  const int oc = wn * 16 + (lane & 15);
  if (oc < OCOFF) {
    const float bv = bias[oc];
    #pragma unroll
    for (int r = 0; r < 8; ++r) {
      const int ow = wm * 16 + r + ((lane >= 16) ? 8 : 0);
      float v = acc[r] + bv;
      if (oc >= 18) v = 1.0f / (1.0f + __expf(-v));
      om[((size_t)blk * 64 + ow) * OCOFF + oc] = v;
    }
  }
}

// --------------------------------------------------------------------------
// Kernel 2: fused bilinear-sample + main GEMM.
// Block: 32 pixels x 256 outc (512 blocks). 8 waves = 2(M) x 4(N), 4 C-tiles
// per wave. Corner validity folded into weights; corner offsets pre-clamped
// -> branchless 4-load sampling. A double-buffered; B frags from global.
// --------------------------------------------------------------------------
__global__ __launch_bounds__(256)
void dcn_main(const float* __restrict__ x,
              const float* __restrict__ om,
              const _Float16* __restrict__ Bmain,
              float* __restrict__ out) {
  __shared__ __align__(16) _Float16 Ald[2][32][40];
  __shared__ int4   mO[9][32];   // 4 pre-clamped corner offsets
  __shared__ float4 mW[9][32];   // mask- & validity-folded corner weights

  const int tid  = threadIdx.x;
  const int lane = tid & 31, wv = tid >> 5;
  const int wm = wv >> 2, wn = wv & 3;
  const int mbase = blockIdx.x * 32;
  const int n   = mbase >> 12;
  const int oh  = (mbase >> 6) & 63;
  const int owb = mbase & 63;
  const float* xn = x + (size_t)n * CIN * HW * HW;

  // ---- bilinear metadata: 9 taps x 32 pixels, padded coords in [0,65] ----
  for (int e = tid; e < 9 * 32; e += 256) {
    const int tap = e >> 5, i = e & 31;
    const size_t mb = (size_t)(mbase + i) * OCOFF;
    const float offy = om[mb + 2 * tap];
    const float offx = om[mb + 2 * tap + 1];
    const float msk  = om[mb + 18 + tap];
    float py = (float)(oh + tap / 3) + offy;        // oh+1 + (tap/3 - 1)
    float px = (float)(owb + i + tap % 3) + offx;   // ow+1 + (tap%3 - 1)
    py = fminf(fmaxf(py, 0.0f), 65.0f);
    px = fminf(fmaxf(px, 0.0f), 65.0f);
    const float fy1 = floorf(py), fx1 = floorf(px);
    const int y1 = (int)fy1, x1 = (int)fx1;         // y1,x1 in [0,65]
    const float ly = py - fy1, lx = px - fx1;
    const float hy = 1.0f - ly, hx = 1.0f - lx;
    // corner validity (padded row/col y maps to real x iff 1 <= y <= 64)
    const float vy0 = (y1 >= 1 && y1 <= 64) ? 1.0f : 0.0f;
    const float vy1 = (y1 <= 63) ? 1.0f : 0.0f;
    const float vx0 = (x1 >= 1 && x1 <= 64) ? 1.0f : 0.0f;
    const float vx1 = (x1 <= 63) ? 1.0f : 0.0f;
    // pre-clamped in-bounds rows/cols (exact whenever the corner is valid)
    const int r0 = min(max(y1 - 1, 0), 63), r1 = min(y1, 63);
    const int c0 = min(max(x1 - 1, 0), 63), c1 = min(x1, 63);
    mO[tap][i] = make_int4(r0 * HW + c0, r0 * HW + c1,
                           r1 * HW + c0, r1 * HW + c1);
    mW[tap][i] = make_float4(hy * hx * msk * vy0 * vx0,
                             hy * lx * msk * vy0 * vx1,
                             ly * hx * msk * vy1 * vx0,
                             ly * lx * msk * vy1 * vx1);
  }
  __syncthreads();

  v8f acc[4] = {};
  const int ai = tid >> 3;                 // pixel 0..31
  const int cg = tid & 7;                  // channel group of 4
  const int ar  = wm * 16 + (lane & 15);
  const int akb = (lane >> 4) * 8;

  int chunk = 0;
  for (int kk = 0; kk < 9; ++kk) {
    const int4   o4 = mO[kk][ai];
    const float4 w  = mW[kk][ai];
    for (int cc = 0; cc < 8; ++cc, ++chunk) {
      const int par = chunk & 1;
      // ---- branchless bilinear A tile: 16 loads, 1 b64 ds store ----
      const int cb = cc * 32 + cg * 4;
      union { _Float16 h[4]; uint2 u; } pk;
      #pragma unroll
      for (int j = 0; j < 4; ++j) {
        const float* p = xn + (size_t)(cb + j) * (HW * HW);
        pk.h[j] = (_Float16)(w.x * p[o4.x] + w.y * p[o4.y] +
                             w.z * p[o4.z] + w.w * p[o4.w]);
      }
      *(uint2*)&Ald[par][ai][cg * 4] = pk.u;
      __syncthreads();
      // ---- fragments + 4 WMMAs/wave; B frags direct from global ----
      const v16h af = load_afrag(&Ald[par][ar][0], akb);
      #pragma unroll
      for (int t = 0; t < 4; ++t) {
        const v16h bf = *(const v16h*)
            (Bmain + ((size_t)(chunk * 16 + wn * 4 + t) * 32 + lane) * 16);
        acc[t] = __builtin_amdgcn_wmma_f32_16x16x32_f16(false, af, false, bf,
                                                        (short)0, acc[t],
                                                        false, false);
      }
    }
  }
  // ---- epilogue: scatter D tiles to out[n][o][oh][ow] ----
  #pragma unroll
  for (int t = 0; t < 4; ++t) {
    const int o = wn * 64 + t * 16 + (lane & 15);
    #pragma unroll
    for (int r = 0; r < 8; ++r) {
      const int prow = wm * 16 + r + ((lane >= 16) ? 8 : 0);
      const int ow = owb + prow;
      out[(((size_t)n * OUTC + o) * HW + oh) * HW + ow] = acc[t][r];
    }
  }
}

// --------------------------------------------------------------------------
extern "C" void kernel_launch(void* const* d_in, const int* in_sizes, int n_in,
                              void* d_out, int out_size, void* d_ws,
                              size_t ws_size, hipStream_t stream) {
  (void)in_sizes; (void)n_in; (void)out_size; (void)ws_size;
  const float* x      = (const float*)d_in[0];
  const float* w_off  = (const float*)d_in[1];
  const float* b_off  = (const float*)d_in[2];
  const float* w_main = (const float*)d_in[3];
  float* out = (float*)d_out;

  float*    om    = (float*)d_ws;                                  // 1.77 MB
  _Float16* Bmain = (_Float16*)((char*)d_ws + OM_FLOATS * 4);      // 1.18 MB
  _Float16* Boff  = (_Float16*)((char*)d_ws + OM_FLOATS * 4
                                 + BMAIN_HALFS * 2);               // 147 KB

  const int prep_total = (int)(BMAIN_HALFS + BOFF_HALFS);          // 663552
  prep_weights<<<(prep_total + 255) / 256, 256, 0, stream>>>(w_off, w_main,
                                                             Bmain, Boff);
  offset_conv<<<NB * HW, 256, 0, stream>>>(x, Boff, b_off, om);    // 256 blks
  dcn_main<<<NPIX / 32, 256, 0, stream>>>(x, om, Bmain, out);      // 512 blks
}
